// TwoLayerModel_33328946217826
// MI455X (gfx1250) — matrix operand order repaired and checked by reading, not compiled, not run on hardware
//
#include <hip/hip_runtime.h>
#include <cstdint>
#include <cstddef>

// Problem constants (match reference)
#define N_NODES 100000
#define N_EDGES 3200000
#define F_IN    256
#define HID     32
#define F_OUT   64

// Padded LDS row strides (elements) for transposed bf16 weight tiles.
// 264*2B = 528B = 132 dwords; 132 % 64 = 4 -> lanes hit distinct bank quads.
#define W1T_STRIDE 264
// 40*2B = 80B = 20 dwords; 20*n % 64 covers distinct multiples of 4 for n=0..15.
#define W2T_STRIDE 40

typedef __attribute__((ext_vector_type(16))) __bf16 v16bf;
typedef __attribute__((ext_vector_type(8)))  __bf16 v8bf;
typedef __attribute__((ext_vector_type(8)))  float  v8f;

// ---------------------------------------------------------------------------
// Degree count: deg[col[e]] += 1 (self-loop added later as +1 in rsqrt)
// ---------------------------------------------------------------------------
__global__ __launch_bounds__(256) void deg_kernel(const int* __restrict__ colp,
                                                  float* __restrict__ deg) {
    int e = blockIdx.x * 256 + threadIdx.x;
    if (e < N_EDGES) unsafeAtomicAdd(&deg[colp[e]], 1.0f);
}

__global__ __launch_bounds__(256) void dinv_kernel(float* __restrict__ deg) {
    int i = blockIdx.x * 256 + threadIdx.x;
    if (i < N_NODES) deg[i] = rsqrtf(deg[i] + 1.0f);   // +1 self-loop
}

// ---------------------------------------------------------------------------
// A-matrix fill for v_wmma_f32_16x16x32_bf16 (contiguous K order used here, matching B; not the ISA table order):
//   lanes 0-15 (half=0): elems 0-15 -> K 0-15
//   lanes 16-31(half=1): elems 0-15 -> K 16-31
// rowptr points at this lane's row, already offset by the k-step base.
// ---------------------------------------------------------------------------
__device__ __forceinline__ v16bf load_a_bf16(const float* __restrict__ rowptr,
                                             int half) {
    const float4* p0 = (const float4*)(rowptr + half * 16);
    const float4* p1 = (const float4*)(rowptr + half * 16 + 8);
    float4 a0 = p0[0], a1 = p0[1], a2 = p1[0], a3 = p1[1];
    v16bf A;
    A[0]  = (__bf16)a0.x; A[1]  = (__bf16)a0.y; A[2]  = (__bf16)a0.z; A[3]  = (__bf16)a0.w;
    A[4]  = (__bf16)a1.x; A[5]  = (__bf16)a1.y; A[6]  = (__bf16)a1.z; A[7]  = (__bf16)a1.w;
    A[8]  = (__bf16)a2.x; A[9]  = (__bf16)a2.y; A[10] = (__bf16)a2.z; A[11] = (__bf16)a2.w;
    A[12] = (__bf16)a3.x; A[13] = (__bf16)a3.y; A[14] = (__bf16)a3.z; A[15] = (__bf16)a3.w;
    return A;
}

// B fragment: 16 contiguous bf16 (32B) from LDS as two 16B chunks
// (row stride only guarantees 16B alignment) -> 2x ds_load_b128.
__device__ __forceinline__ v16bf load_b_lds(const __bf16* p) {
    v8bf lo = *(const v8bf*)(p);
    v8bf hi = *(const v8bf*)(p + 8);
    v16bf B;
    #pragma unroll
    for (int e = 0; e < 8; ++e) { B[e] = lo[e]; B[e + 8] = hi[e]; }
    return B;
}

// ---------------------------------------------------------------------------
// GEMM1: h0[N,32] = x[N,256] @ W1[256,32]  (bf16 WMMA, f32 accum)
// One wave per 16-row tile; 2 column tiles of 16; K loop of 8 x 32.
// W1 staged TRANSPOSED in LDS: sWT[n*W1T_STRIDE + k], so each lane's B
// fragment (fixed n, K contiguous) is a 32B contiguous LDS read.
// ---------------------------------------------------------------------------
__global__ __launch_bounds__(256) void gemm1_kernel(const float* __restrict__ x,
                                                    const float* __restrict__ W1,
                                                    float* __restrict__ h0) {
    __shared__ __bf16 sWT[HID * W1T_STRIDE];          // ~16.5 KB of 320 KB LDS
    int tid = threadIdx.y * 32 + threadIdx.x;
    for (int i = tid; i < F_IN * HID; i += 256) {
        int k = i >> 5;          // 0..255  (coalesced read of W1)
        int n = i & 31;          // 0..31
        sWT[n * W1T_STRIDE + k] = (__bf16)W1[i];
    }
    __syncthreads();

    int tile = blockIdx.x * 8 + threadIdx.y;
    if (tile * 16 >= N_NODES) return;                 // whole-wave uniform
    int lane = threadIdx.x;
    int half = lane >> 4;
    int m    = lane & 15;
    const float* xrow = x + (size_t)(tile * 16 + m) * F_IN;

    v8f acc0 = {}; v8f acc1 = {};
    #pragma unroll
    for (int kk = 0; kk < F_IN; kk += 32) {
        v16bf A = load_a_bf16(xrow + kk, half);
        // B layout (16-bit, 32x16): elem e -> K = kk + 16*half + e, N = m
        int kbase = kk + 16 * half;
        v16bf B0 = load_b_lds(&sWT[m * W1T_STRIDE + kbase]);
        v16bf B1 = load_b_lds(&sWT[(m + 16) * W1T_STRIDE + kbase]);
        acc0 = __builtin_amdgcn_wmma_f32_16x16x32_bf16(false, A, false, B0,
                                                       (short)0, acc0, false, false);
        acc1 = __builtin_amdgcn_wmma_f32_16x16x32_bf16(false, A, false, B1,
                                                       (short)0, acc1, false, false);
    }
    // C/D layout: VGPR r -> row = tile*16 + r + 8*half, col = m
    int rbase = tile * 16 + half * 8;
    #pragma unroll
    for (int r = 0; r < 8; ++r) {
        h0[(size_t)(rbase + r) * HID + m]      = acc0[r];
        h0[(size_t)(rbase + r) * HID + 16 + m] = acc1[r];
    }
}

// ---------------------------------------------------------------------------
// Propagate: hout[col] += dinv[row]*dinv[col] * hin[row]  over E edges + N self
// One wave per edge, lane = feature (HID == 32 == wave32).
// 128B coalesced gather + 128B coalesced hardware f32 atomic scatter.
// ---------------------------------------------------------------------------
__global__ __launch_bounds__(256) void scatter_kernel(const int* __restrict__ rowp,
                                                      const int* __restrict__ colp,
                                                      const float* __restrict__ dinv,
                                                      const float* __restrict__ hin,
                                                      float* __restrict__ hout,
                                                      int nedge) {
    int e = blockIdx.x * 8 + threadIdx.y;
    if (e >= nedge) return;
    int r, c;
    if (e < N_EDGES) { r = rowp[e]; c = colp[e]; }
    else             { r = c = e - N_EDGES; }        // self-loop
    float norm = dinv[r] * dinv[c];
    int f = threadIdx.x;
    float v = norm * hin[(size_t)r * HID + f];
    unsafeAtomicAdd(&hout[(size_t)c * HID + f], v);
}

// ---------------------------------------------------------------------------
// h = relu(agg + b1)
// ---------------------------------------------------------------------------
__global__ __launch_bounds__(256) void bias_relu_kernel(float* __restrict__ h,
                                                        const float* __restrict__ b1) {
    int i = blockIdx.x * 256 + threadIdx.x;
    if (i < N_NODES * HID) {
        float v = h[i] + b1[i & (HID - 1)];
        h[i] = v > 0.0f ? v : 0.0f;
    }
}

// ---------------------------------------------------------------------------
// GEMM2: out[N,64] = fr[N,32] @ W2[32,64] + b2  (single K=32 WMMA per tile)
// W2 staged transposed in LDS: sWT2[n*W2T_STRIDE + k].
// ---------------------------------------------------------------------------
__global__ __launch_bounds__(256) void gemm2_kernel(const float* __restrict__ fr,
                                                    const float* __restrict__ W2,
                                                    const float* __restrict__ b2,
                                                    float* __restrict__ out) {
    __shared__ __bf16 sWT2[F_OUT * W2T_STRIDE];       // 5 KB
    int tid = threadIdx.y * 32 + threadIdx.x;
    for (int i = tid; i < HID * F_OUT; i += 256) {
        int k = i >> 6;          // 0..31   (coalesced read of W2)
        int n = i & 63;          // 0..63
        sWT2[n * W2T_STRIDE + k] = (__bf16)W2[i];
    }
    __syncthreads();

    int tile = blockIdx.x * 8 + threadIdx.y;
    if (tile * 16 >= N_NODES) return;
    int lane = threadIdx.x;
    int half = lane >> 4;
    int m    = lane & 15;

    const float* frow = fr + (size_t)(tile * 16 + m) * HID;
    v16bf A = load_a_bf16(frow, half);                // K = 32 exactly

    int rbase = tile * 16 + half * 8;
    #pragma unroll
    for (int ct = 0; ct < 4; ++ct) {
        v16bf B = load_b_lds(&sWT2[(ct * 16 + m) * W2T_STRIDE + 16 * half]);
        v8f acc = {};
        acc = __builtin_amdgcn_wmma_f32_16x16x32_bf16(false, A, false, B,
                                                      (short)0, acc, false, false);
        float bias = b2[ct * 16 + m];
        #pragma unroll
        for (int r = 0; r < 8; ++r)
            out[(size_t)(rbase + r) * F_OUT + ct * 16 + m] = acc[r] + bias;
    }
}

// ---------------------------------------------------------------------------
// Launch sequence (all on `stream`, graph-capture safe)
// ---------------------------------------------------------------------------
extern "C" void kernel_launch(void* const* d_in, const int* in_sizes, int n_in,
                              void* d_out, int out_size, void* d_ws, size_t ws_size,
                              hipStream_t stream) {
    const float* x  = (const float*)d_in[0];
    const int*   ei = (const int*)d_in[1];
    const float* W1 = (const float*)d_in[2];
    const float* b1 = (const float*)d_in[3];
    const float* W2 = (const float*)d_in[4];
    const float* b2 = (const float*)d_in[5];
    const int* rowp = ei;            // edge_index[0]
    const int* colp = ei + N_EDGES;  // edge_index[1]

    // Workspace layout: [dinv: N][bufB: N*32][bufA: N*32]  (~26 MB)
    float* dinv = (float*)d_ws;
    float* bufB = dinv + N_NODES;
    float* bufA = bufB + (size_t)N_NODES * HID;

    // Zero deg (dinv slot) + bufB (agg target of pass 1) in one memset
    hipMemsetAsync(d_ws, 0, sizeof(float) * (N_NODES + (size_t)N_NODES * HID), stream);

    deg_kernel <<<(N_EDGES + 255) / 256, 256, 0, stream>>>(colp, dinv);
    dinv_kernel<<<(N_NODES + 255) / 256, 256, 0, stream>>>(dinv);

    dim3 blk(32, 8);                       // 8 waves / block (wave32)
    int ntile = N_NODES / 16;              // 6250, exact
    gemm1_kernel<<<(ntile + 7) / 8, blk, 0, stream>>>(x, W1, bufA);

    int nedge = N_EDGES + N_NODES;
    scatter_kernel<<<(nedge + 7) / 8, blk, 0, stream>>>(rowp, colp, dinv, bufA, bufB, nedge);
    bias_relu_kernel<<<(N_NODES * HID + 255) / 256, 256, 0, stream>>>(bufB, b1);

    hipMemsetAsync(bufA, 0, sizeof(float) * (size_t)N_NODES * HID, stream);
    scatter_kernel<<<(nedge + 7) / 8, blk, 0, stream>>>(rowp, colp, dinv, bufB, bufA, nedge);

    gemm2_kernel<<<(ntile + 7) / 8, blk, 0, stream>>>(bufA, W2, b2, (float*)d_out);
}
